// PolyNetFP4_76355928588537
// MI455X (gfx1250) — matrix operand order repaired and checked by reading, not compiled, not run on hardware
//
#include <hip/hip_runtime.h>
#include <hip/hip_bf16.h>

typedef __attribute__((ext_vector_type(16))) _Float16 v16h;
typedef __attribute__((ext_vector_type(8)))  float    v8f;
typedef __attribute__((ext_vector_type(8)))  unsigned v8u;
typedef __attribute__((ext_vector_type(2)))  __fp16   v2hf;   // cvt_pkrtz result type

// f32 region of ws (word offsets)
#define WS_F32_W1 0
#define WS_F32_B1 64
#define WS_F32_B2 128
#define WS_F32_B3 192
#define WS_F32_B4 224
// f16 region starts at byte 1024; offsets below in v16h (32B) units
#define WS_V16_W2A 0     // 8 tiles * 32 lanes
#define WS_V16_W3A 256   // 4 tiles * 32
#define WS_V16_W4A 384   // 1 tile  * 32

__device__ __forceinline__ float fp4_nearest(float v) {
    const float codes[16] = {0.0f, 0.0052083333f, 0.6666667f, 1.0f,
                             0.33333334f, 0.5f, 0.16666667f, 0.25f,
                             -0.0f, -0.0052083333f, -0.6666667f, -1.0f,
                             -0.33333334f, -0.5f, -0.16666667f, -0.25f};
    float best = codes[0];
    float bd = fabsf(v - codes[0]);
#pragma unroll
    for (int i = 1; i < 16; ++i) {
        float d = fabsf(v - codes[i]);
        if (d < bd) { bd = d; best = codes[i]; }
    }
    return best;
}

// Input is t = 0.5*preactivation (0.5 folded into weights/biases of layers 1-3).
// silu(preact) = t + t*tanh(t)  ->  v_tanh_f32 + v_fma : 2 VALU ops per value.
__device__ __forceinline__ float silu_t(float t) {
    return fmaf(t, __builtin_amdgcn_tanhf(t), t);
}

__device__ __forceinline__ unsigned pack_silu_pair(float ta, float tb) {
    v2hf h = __builtin_amdgcn_cvt_pkrtz(silu_t(ta), silu_t(tb));
    return __builtin_bit_cast(unsigned, h);
}

// ---------------------------------------------------------------------------
// Prep: FP4 roundtrip of w1..w4 (blocksize-64 absmax, nearest code), then pack
// W2/W3/W4 into WMMA A-operand f16 lane layout + f32 w1/biases into d_ws.
// Layers 1-3 weights/biases are scaled by 0.5 AFTER quantization (tanh trick).
// A-layout (16x32 f16): lane L holds row M=L%16; element e <-> K =
//   e + 16*(L/16)  (+32*ktile).
// ---------------------------------------------------------------------------
__global__ void __launch_bounds__(256) fp4mlp_prep(
    const float* __restrict__ w1, const float* __restrict__ b1,
    const float* __restrict__ w2, const float* __restrict__ b2,
    const float* __restrict__ w3, const float* __restrict__ b3,
    const float* __restrict__ w4, const float* __restrict__ b4,
    unsigned char* __restrict__ ws)
{
    __shared__ float s[6240]; // [0,64) w1q | [64,4160) w2q | [4160,6208) w3q | [6208,6240) w4q
    const int tid = threadIdx.x;

    for (int b = tid; b < 98; b += 256) {
        const float* src; float* dst; int cnt;
        if (b == 0)      { src = w1;               dst = s;                 cnt = 64; }
        else if (b < 65) { src = w2 + (b - 1) * 64;  dst = s + 64 + (b - 1) * 64;   cnt = 64; }
        else if (b < 97) { src = w3 + (b - 65) * 64; dst = s + 4160 + (b - 65) * 64; cnt = 64; }
        else             { src = w4;               dst = s + 6208;          cnt = 32; }
        float am = 0.0f;
        for (int i = 0; i < cnt; ++i) am = fmaxf(am, fabsf(src[i]));
        float sc = (am == 0.0f) ? 1.0f : am;
        for (int i = 0; i < cnt; ++i) dst[i] = fp4_nearest(src[i] / sc) * am;
    }
    __syncthreads();

    float* wf = (float*)ws;
    for (int i = tid; i < 64; i += 256) {
        wf[WS_F32_W1 + i] = 0.5f * s[i];       // fold tanh half-scale
        wf[WS_F32_B1 + i] = 0.5f * b1[i];
        wf[WS_F32_B2 + i] = 0.5f * b2[i];
    }
    for (int i = tid; i < 32; i += 256) wf[WS_F32_B3 + i] = 0.5f * b3[i];
    if (tid == 0) wf[WS_F32_B4] = b4[0];       // layer 4 unscaled

    _Float16* hpk = (_Float16*)(ws + 1024);
    // W2 A-layout (x0.5): 4 Mtiles (n) x 2 Ktiles (j): A[n][j] = 0.5*w2q[n][j]
    for (int idx = tid; idx < 4096; idx += 256) {
        int tile = idx >> 9, lane = (idx >> 4) & 31, e = idx & 15;
        int mt = tile >> 1, kt = tile & 1;
        int nn = 16 * mt + (lane & 15);
        int j  = 32 * kt + e + 16 * (lane >> 4);
        hpk[idx] = (_Float16)(0.5f * s[64 + nn * 64 + j]);
    }
    // W3 A-layout (x0.5): 2 Mtiles (p) x 2 Ktiles (n): A[p][n] = 0.5*w3q[p][n]
    for (int idx = tid; idx < 2048; idx += 256) {
        int tile = idx >> 9, lane = (idx >> 4) & 31, e = idx & 15;
        int mt = tile >> 1, kt = tile & 1;
        int pp = 16 * mt + (lane & 15);
        int j  = 32 * kt + e + 16 * (lane >> 4);
        hpk[4096 + idx] = (_Float16)(0.5f * s[4160 + pp * 64 + j]);
    }
    // W4 A-layout (unscaled): row 0 = w4q[k], rows 1..15 = 0
    for (int idx = tid; idx < 512; idx += 256) {
        int lane = idx >> 4, e = idx & 15;
        int k = e + 16 * (lane >> 4);
        hpk[6144 + idx] = ((lane & 15) == 0) ? (_Float16)s[6208 + k] : (_Float16)0.0f;
    }
}

// ---------------------------------------------------------------------------
// Main: transposed dataflow h^T = W * h_prev^T so each layer's D (C-layout)
// converts to the next layer's B operand via f16 pack + ds_swizzle SWAPX16.
// 13 v_wmma_f32_16x16x32_f16 per 16-element tile.
// ---------------------------------------------------------------------------
__global__ void __launch_bounds__(256) fp4mlp_main(
    const float* __restrict__ x, float* __restrict__ out,
    const unsigned char* __restrict__ ws,
    int n, int nTiles, int totalWaves)
{
    const int lane = threadIdx.x & 31;
    const int hi   = lane >> 4;      // which 16-lane half
    const int m    = lane & 15;      // batch column this lane owns in B/D tiles
    const int wave = (int)((blockIdx.x * blockDim.x + threadIdx.x) >> 5);

    const float* wf = (const float*)ws;
    const v16h*  hp = (const v16h*)(ws + 1024);

    // Resident weights in A-operand layout
    v16h W2r[4][2], W3r[2][2], W4r;
#pragma unroll
    for (int mt = 0; mt < 4; ++mt)
#pragma unroll
        for (int kt = 0; kt < 2; ++kt)
            W2r[mt][kt] = hp[WS_V16_W2A + (mt * 2 + kt) * 32 + lane];
#pragma unroll
    for (int mt = 0; mt < 2; ++mt)
#pragma unroll
        for (int kt = 0; kt < 2; ++kt)
            W3r[mt][kt] = hp[WS_V16_W3A + (mt * 2 + kt) * 32 + lane];
    W4r = hp[WS_V16_W4A + lane];

    // Layer-1 weights/bias (pre-scaled by 0.5), indexed to land directly in
    // B-operand layout: element c of Bh[kt]: j = 32*kt + 16*hi + (c&15)
    float w1r[32], b1r[32];
#pragma unroll
    for (int c = 0; c < 32; ++c) {
        int j = 32 * (c >> 4) + 16 * hi + (c & 15);
        w1r[c] = wf[WS_F32_W1 + j];
        b1r[c] = wf[WS_F32_B1 + j];
    }
    // Per-lane biases (pre-scaled) in C/D-layout rows: row = 16*mt + 8*hi + r
    float b2A[32];
#pragma unroll
    for (int mt = 0; mt < 4; ++mt)
#pragma unroll
        for (int r = 0; r < 8; ++r)
            b2A[mt * 8 + r] = wf[WS_F32_B2 + 16 * mt + 8 * hi + r];
    float b3A[16];
#pragma unroll
    for (int mt = 0; mt < 2; ++mt)
#pragma unroll
        for (int r = 0; r < 8; ++r)
            b3A[mt * 8 + r] = wf[WS_F32_B3 + 16 * mt + 8 * hi + r];
    const float b4v = wf[WS_F32_B4];

    for (int t = wave; t < nTiles; t += totalWaves) {
        const long base = (long)t * 16;
        int xi = (int)base + m;
        if (xi >= n) xi = n - 1;           // clamp, keep EXEC full
        const float xv = x[xi];

        // ---- Layer 1 (VALU) directly into B-operand layout: h1^T[j][m] ----
        v16h Bh[2];
#pragma unroll
        for (int c = 0; c < 32; ++c) {
            float tpre = fmaf(xv, w1r[c], b1r[c]);     // already 0.5*preact
            Bh[c >> 4][c & 15] = (_Float16)silu_t(tpre);
        }

        // ---- Layer 2: t2^T = 0.5*W2 x h1^T  (4 Mtiles x 2 Ktiles = 8 WMMA) -
        v8f D2[4];
#pragma unroll
        for (int mt = 0; mt < 4; ++mt) {
            v8f acc = {};
            acc = __builtin_amdgcn_wmma_f32_16x16x32_f16(false, W2r[mt][0], false, Bh[0],
                                                         (short)0, acc, false, false);
            acc = __builtin_amdgcn_wmma_f32_16x16x32_f16(false, W2r[mt][1], false, Bh[1],
                                                         (short)0, acc, false, false);
            D2[mt] = acc;
        }

        // bias + silu + pack to f16 pairs; swizzle for C->B layout conversion
        unsigned P[4][4], SW[4][4];
#pragma unroll
        for (int mt = 0; mt < 4; ++mt) {
#pragma unroll
            for (int d = 0; d < 4; ++d) {
                P[mt][d]  = pack_silu_pair(D2[mt][2 * d]     + b2A[mt * 8 + 2 * d],
                                           D2[mt][2 * d + 1] + b2A[mt * 8 + 2 * d + 1]);
                SW[mt][d] = (unsigned)__builtin_amdgcn_ds_swizzle((int)P[mt][d], 0x401f); // SWAPX16
            }
        }
        v16h B2h[2];
#pragma unroll
        for (int kt = 0; kt < 2; ++kt) {
            v8u bw;
#pragma unroll
            for (int d = 0; d < 4; ++d) {
                bw[d]     = hi ? SW[2 * kt + 1][d] : P[2 * kt][d];
                bw[d + 4] = hi ? P[2 * kt + 1][d]  : SW[2 * kt][d];
            }
            B2h[kt] = __builtin_bit_cast(v16h, bw);
        }

        // ---- Layer 3: t3^T = 0.5*W3 x h2^T  (2 Mtiles x 2 Ktiles = 4 WMMA) -
        v8f D3[2];
#pragma unroll
        for (int mt = 0; mt < 2; ++mt) {
            v8f acc = {};
            acc = __builtin_amdgcn_wmma_f32_16x16x32_f16(false, W3r[mt][0], false, B2h[0],
                                                         (short)0, acc, false, false);
            acc = __builtin_amdgcn_wmma_f32_16x16x32_f16(false, W3r[mt][1], false, B2h[1],
                                                         (short)0, acc, false, false);
            D3[mt] = acc;
        }
        unsigned Q[2][4], SQ[2][4];
#pragma unroll
        for (int mt = 0; mt < 2; ++mt) {
#pragma unroll
            for (int d = 0; d < 4; ++d) {
                Q[mt][d]  = pack_silu_pair(D3[mt][2 * d]     + b3A[mt * 8 + 2 * d],
                                           D3[mt][2 * d + 1] + b3A[mt * 8 + 2 * d + 1]);
                SQ[mt][d] = (unsigned)__builtin_amdgcn_ds_swizzle((int)Q[mt][d], 0x401f);
            }
        }
        v8u bw4;
#pragma unroll
        for (int d = 0; d < 4; ++d) {
            bw4[d]     = hi ? SQ[1][d] : Q[0][d];
            bw4[d + 4] = hi ? Q[1][d]  : SQ[0][d];
        }
        v16h B3h = __builtin_bit_cast(v16h, bw4);

        // ---- Layer 4: out^T = w4 x h3^T (1 WMMA, row 0 of D is the output) --
        v8f z = {};
        v8f D4 = __builtin_amdgcn_wmma_f32_16x16x32_f16(false, W4r, false, B3h,
                                                        (short)0, z, false, false);

        const long oi = base + lane;
        if (lane < 16 && oi < (long)n) out[oi] = D4[0] + b4v;
    }
}

extern "C" void kernel_launch(void* const* d_in, const int* in_sizes, int n_in,
                              void* d_out, int out_size, void* d_ws, size_t ws_size,
                              hipStream_t stream)
{
    const float* x  = (const float*)d_in[0];
    const float* w1 = (const float*)d_in[1];
    const float* b1 = (const float*)d_in[2];
    const float* w2 = (const float*)d_in[3];
    const float* b2 = (const float*)d_in[4];
    const float* w3 = (const float*)d_in[5];
    const float* b3 = (const float*)d_in[6];
    const float* w4 = (const float*)d_in[7];
    const float* b4 = (const float*)d_in[8];
    float* out = (float*)d_out;
    const int n = in_sizes[0];

    fp4mlp_prep<<<1, 256, 0, stream>>>(w1, b1, w2, b2, w3, b3, w4, b4,
                                       (unsigned char*)d_ws);

    const int nTiles = (n + 15) >> 4;
    int blocks = (nTiles + 7) / 8;   // 8 waves per 256-thread block
    if (blocks > 2048) blocks = 2048;
    const int totalWaves = blocks * 8;
    fp4mlp_main<<<blocks, 256, 0, stream>>>(x, out, (const unsigned char*)d_ws,
                                            n, nTiles, totalWaves);
}
